// TriangleAttention_47914655154274
// MI455X (gfx1250) — matrix operand order repaired and checked by reading, not compiled, not run on hardware
//
#include <hip/hip_runtime.h>
#include <hip/hip_bf16.h>

typedef __attribute__((ext_vector_type(16))) _Float16 v16h;
typedef __attribute__((ext_vector_type(8)))  _Float16 v8h;
typedef __attribute__((ext_vector_type(8)))  float    v8f;

#define NSEQ   320
#define NPAIR  (NSEQ * NSEQ)      // 102400
#define DIMD   128
#define HEADS  4
#define DHEAD  64
#define INNER  256
#define QKVN   768

__device__ __forceinline__ v8f wmma_f16(v16h a, v16h b, v8f c) {
  return __builtin_amdgcn_wmma_f32_16x16x32_f16(false, a, false, b, (short)0, c, false, false);
}

// A fragment (16x32, f16 source): halves 0..7 = p[0..7], halves 8..15 = p[16..23]
__device__ __forceinline__ v16h load_a16(const _Float16* p) {
  v8h x = *(const v8h*)p;
  v8h y = *(const v8h*)(p + 16);
  v16h r;
#pragma unroll
  for (int e = 0; e < 8; ++e) { r[e] = x[e]; r[8 + e] = y[e]; }
  return r;
}

// A fragment built from fp32 source (convert on the fly)
__device__ __forceinline__ v16h cvt_load_a(const float* p) {
  v16h r;
#pragma unroll
  for (int e = 0; e < 8; ++e) { r[e] = (_Float16)p[e]; r[8 + e] = (_Float16)p[16 + e]; }
  return r;
}

// B fragment (32x16, column-major source): halves 0..7 = p[0..7], halves 8..15 = p[16..23]
__device__ __forceinline__ v16h load_b16(const _Float16* p) {
  v8h x = *(const v8h*)p;
  v8h y = *(const v8h*)(p + 16);
  v16h r;
#pragma unroll
  for (int e = 0; e < 8; ++e) { r[e] = x[e]; r[8 + e] = y[e]; }
  return r;
}

// ---------------------------------------------------------------------------
// K0: transpose + f16-convert weights: Wqkv[128,768] -> Wqkvt[768,128],
//     Wout[256,128] -> Woutt[128,256]
// ---------------------------------------------------------------------------
__global__ void prep_weights(const float* __restrict__ Wqkv,
                             const float* __restrict__ Wout,
                             _Float16* __restrict__ Wqkvt,
                             _Float16* __restrict__ Woutt) {
  int idx = blockIdx.x * blockDim.x + threadIdx.x;
  if (idx < QKVN * DIMD) {
    int n = idx >> 7, kk = idx & 127;
    Wqkvt[idx] = (_Float16)Wqkv[kk * QKVN + n];
  } else {
    int idx2 = idx - QKVN * DIMD;
    if (idx2 < DIMD * INNER) {
      int n = idx2 >> 8, kk = idx2 & 255;
      Woutt[idx2] = (_Float16)Wout[kk * DIMD + n];
    }
  }
}

// ---------------------------------------------------------------------------
// K1: attn_bias[h,q,k] = sum_d P[q,k,d] * Wbias[d,h]   (fp32, tiny GEMV)
// ---------------------------------------------------------------------------
__global__ void __launch_bounds__(256) bias_kernel(const float* __restrict__ P,
                                                   const float* __restrict__ Wbias,
                                                   float* __restrict__ bias) {
  __shared__ float wb[DIMD * HEADS];
  int tid = threadIdx.x;
  wb[tid] = Wbias[tid];
  wb[tid + 256] = Wbias[tid + 256];
  __syncthreads();
  int p = blockIdx.x * 256 + tid;        // pair index (q,k)
  const float* pp = P + (size_t)p * DIMD;
  float a0 = 0.f, a1 = 0.f, a2 = 0.f, a3 = 0.f;
#pragma unroll 8
  for (int d = 0; d < DIMD; ++d) {
    float x = pp[d];
    a0 += x * wb[d * 4 + 0];
    a1 += x * wb[d * 4 + 1];
    a2 += x * wb[d * 4 + 2];
    a3 += x * wb[d * 4 + 3];
  }
  int q = p / NSEQ, k = p - q * NSEQ;
  size_t o = (size_t)q * NSEQ + k;
  bias[o] = a0;
  bias[(size_t)NPAIR + o] = a1;
  bias[(size_t)2 * NPAIR + o] = a2;
  bias[(size_t)3 * NPAIR + o] = a3;
}

// ---------------------------------------------------------------------------
// K2: QKV projection. P[102400,128] (f32) @ Wqkvt'[768,128] -> q/k/vT (f16)
// Block: 256 thr = 8 waves, tile BM=64 x BN=128. Wave: 16x64 (4 WMMA tiles).
// q,k stored [i,h,j,d]; v stored TRANSPOSED [i,h,d,j] for the P@V GEMM.
// ---------------------------------------------------------------------------
__global__ void __launch_bounds__(256) qkv_kernel(const float* __restrict__ P,
                                                  const _Float16* __restrict__ Wt,
                                                  _Float16* __restrict__ qws,
                                                  _Float16* __restrict__ kws,
                                                  _Float16* __restrict__ vtws) {
  int tid = threadIdx.x, wave = tid >> 5, lane = tid & 31;
  int lr = lane & 15, hs = lane >> 4;
  int waveM = wave & 3, waveN = wave >> 2;
  int Mbase = blockIdx.x * 64 + waveM * 16;
  int Nbase = blockIdx.y * 128 + waveN * 64;

  v8f acc[4];
#pragma unroll
  for (int t = 0; t < 4; ++t) acc[t] = (v8f){};

  const float* ap0 = P + (size_t)(Mbase + lr) * DIMD + hs * 8;
#pragma unroll
  for (int kb = 0; kb < DIMD; kb += 32) {
    v16h a = cvt_load_a(ap0 + kb);
#pragma unroll
    for (int nt = 0; nt < 4; ++nt) {
      const _Float16* bp = Wt + (size_t)(Nbase + nt * 16 + lr) * DIMD + kb + hs * 8;
      acc[nt] = wmma_f16(a, load_b16(bp), acc[nt]);
    }
  }

#pragma unroll
  for (int nt = 0; nt < 4; ++nt) {
    int n = Nbase + nt * 16 + lr;
#pragma unroll
    for (int r = 0; r < 8; ++r) {
      int mm = Mbase + r + 8 * hs;
      int i = mm / NSEQ, j = mm - i * NSEQ;
      _Float16 v = (_Float16)acc[nt][r];
      if (n < INNER) {                     // Q
        int hh = n >> 6, d = n & 63;
        qws[(((size_t)i * HEADS + hh) * NSEQ + j) * DHEAD + d] = v;
      } else if (n < 2 * INNER) {          // K
        int n2 = n - INNER, hh = n2 >> 6, d = n2 & 63;
        kws[(((size_t)i * HEADS + hh) * NSEQ + j) * DHEAD + d] = v;
      } else {                             // V, transposed [i,h,d,j]
        int n2 = n - 2 * INNER, hh = n2 >> 6, d = n2 & 63;
        vtws[(((size_t)i * HEADS + hh) * DHEAD + d) * NSEQ + j] = v;
      }
    }
  }
}

// ---------------------------------------------------------------------------
// K3: attention. One wave per (i, h, q-tile of 16).
// scores[16,320] via 40 WMMA; fp32 softmax (shfl reduce over 16 lanes);
// probs -> LDS (f16) -> A fragments; out[16,64] via 40 WMMA against vT.
// ---------------------------------------------------------------------------
#define PROB_STRIDE 336   // 320 + pad, 16B-aligned rows
__global__ void __launch_bounds__(128) attn_kernel(const _Float16* __restrict__ qws,
                                                   const _Float16* __restrict__ kws,
                                                   const _Float16* __restrict__ vtws,
                                                   const float* __restrict__ bias,
                                                   _Float16* __restrict__ aout) {
  __shared__ _Float16 lds[4][16][PROB_STRIDE];
  int tid = threadIdx.x, wave = tid >> 5, lane = tid & 31;
  int lr = lane & 15, hs = lane >> 4;
  int task = blockIdx.x * 4 + wave;          // 0..25599
  int qt = task % 20;
  int ih = task / 20;
  int h = ih & 3, i = ih >> 2;
  int qbase = qt * 16;

  const _Float16* Q  = qws  + ((size_t)i * HEADS + h) * NSEQ * DHEAD;
  const _Float16* Kp = kws  + ((size_t)i * HEADS + h) * NSEQ * DHEAD;
  const _Float16* VT = vtws + ((size_t)i * HEADS + h) * DHEAD * NSEQ;

  // ---- scores = Q @ K^T ----
  v8f sc[20];
#pragma unroll
  for (int t = 0; t < 20; ++t) sc[t] = (v8f){};
#pragma unroll
  for (int kb = 0; kb < DHEAD; kb += 32) {
    v16h a = load_a16(Q + (size_t)(qbase + lr) * DHEAD + kb + hs * 8);
#pragma unroll
    for (int t = 0; t < 20; ++t) {
      v16h b = load_b16(Kp + (size_t)(t * 16 + lr) * DHEAD + kb + hs * 8);
      sc[t] = wmma_f16(a, b, sc[t]);
    }
  }

  // ---- scale + bias ----
  const float scale = 0.125f;  // 64^-0.5
  const float* bp = bias + ((size_t)h * NSEQ + qbase) * NSEQ;
#pragma unroll
  for (int t = 0; t < 20; ++t)
#pragma unroll
    for (int r = 0; r < 8; ++r)
      sc[t][r] = sc[t][r] * scale + bp[(size_t)(r + 8 * hs) * NSEQ + t * 16 + lr];

  // ---- softmax over k (16 lanes x 20 tiles) ----
  float mx[8], sm[8];
#pragma unroll
  for (int r = 0; r < 8; ++r) {
    float m = sc[0][r];
#pragma unroll
    for (int t = 1; t < 20; ++t) m = fmaxf(m, sc[t][r]);
#pragma unroll
    for (int d = 1; d < 16; d <<= 1) m = fmaxf(m, __shfl_xor(m, d, 32));
    mx[r] = m;
  }
#pragma unroll
  for (int r = 0; r < 8; ++r) {
    float s = 0.f;
#pragma unroll
    for (int t = 0; t < 20; ++t) {
      float e = __expf(sc[t][r] - mx[r]);
      sc[t][r] = e;
      s += e;
    }
#pragma unroll
    for (int d = 1; d < 16; d <<= 1) s += __shfl_xor(s, d, 32);
    sm[r] = 1.f / s;
  }

  // ---- probs (f16) -> LDS in row-major [q(16), k(320)] ----
#pragma unroll
  for (int t = 0; t < 20; ++t)
#pragma unroll
    for (int r = 0; r < 8; ++r)
      lds[wave][r + 8 * hs][t * 16 + lr] = (_Float16)(sc[t][r] * sm[r]);
  __syncthreads();

  // ---- out = probs @ V  (B operand from vT, contiguous) ----
  v8f oc[4];
#pragma unroll
  for (int dt = 0; dt < 4; ++dt) oc[dt] = (v8f){};
#pragma unroll
  for (int kc = 0; kc < 10; ++kc) {
    v16h a = load_a16(&lds[wave][lr][kc * 32 + hs * 8]);
#pragma unroll
    for (int dt = 0; dt < 4; ++dt) {
      v16h b = load_b16(VT + (size_t)(dt * 16 + lr) * NSEQ + kc * 32 + hs * 8);
      oc[dt] = wmma_f16(a, b, oc[dt]);
    }
  }

  // ---- store to attn_out [i, q, h*64+d] (f16, inner-major for final GEMM) ----
  _Float16* op = aout + ((size_t)i * NSEQ + qbase) * INNER + h * DHEAD;
#pragma unroll
  for (int dt = 0; dt < 4; ++dt)
#pragma unroll
    for (int r = 0; r < 8; ++r)
      op[(size_t)(r + 8 * hs) * INNER + dt * 16 + lr] = (_Float16)oc[dt][r];
}

// ---------------------------------------------------------------------------
// K4: output projection. AO[102400,256] (f16) @ Woutt'[128,256] + bout -> f32
// Block: 256 thr = 8 waves, BM=64 x BN=128 (= full N).
// ---------------------------------------------------------------------------
__global__ void __launch_bounds__(256) outproj_kernel(const _Float16* __restrict__ AO,
                                                      const _Float16* __restrict__ Wt,
                                                      const float* __restrict__ bout,
                                                      float* __restrict__ out) {
  int tid = threadIdx.x, wave = tid >> 5, lane = tid & 31;
  int lr = lane & 15, hs = lane >> 4;
  int waveM = wave & 3, waveN = wave >> 2;
  int Mbase = blockIdx.x * 64 + waveM * 16;
  int Nbase = waveN * 64;

  v8f acc[4];
#pragma unroll
  for (int t = 0; t < 4; ++t) acc[t] = (v8f){};

  const _Float16* ap0 = AO + (size_t)(Mbase + lr) * INNER + hs * 8;
#pragma unroll
  for (int kb = 0; kb < INNER; kb += 32) {
    v16h a = load_a16(ap0 + kb);
#pragma unroll
    for (int nt = 0; nt < 4; ++nt) {
      const _Float16* bpt = Wt + (size_t)(Nbase + nt * 16 + lr) * INNER + kb + hs * 8;
      acc[nt] = wmma_f16(a, load_b16(bpt), acc[nt]);
    }
  }

#pragma unroll
  for (int nt = 0; nt < 4; ++nt) {
    int n = Nbase + nt * 16 + lr;
    float bb = bout[n];
#pragma unroll
    for (int r = 0; r < 8; ++r) {
      int mm = Mbase + r + 8 * hs;
      out[(size_t)mm * DIMD + n] = acc[nt][r] + bb;
    }
  }
}

// ---------------------------------------------------------------------------
extern "C" void kernel_launch(void* const* d_in, const int* in_sizes, int n_in,
                              void* d_out, int out_size, void* d_ws, size_t ws_size,
                              hipStream_t stream) {
  (void)in_sizes; (void)n_in; (void)out_size; (void)ws_size;
  const float* P     = (const float*)d_in[0];   // [1,320,320,128]
  const float* Wqkv  = (const float*)d_in[1];   // [128,768]
  const float* Wout  = (const float*)d_in[2];   // [256,128]
  const float* bout  = (const float*)d_in[3];   // [128]
  const float* Wbias = (const float*)d_in[4];   // [128,4]
  float* out = (float*)d_out;

  char* ws = (char*)d_ws;
  const size_t sz_qkv = (size_t)NSEQ * HEADS * NSEQ * DHEAD * sizeof(_Float16); // 52.4MB each
  _Float16* qws   = (_Float16*)(ws);
  _Float16* kws   = (_Float16*)(ws + sz_qkv);
  _Float16* vtws  = (_Float16*)(ws + 2 * sz_qkv);
  float*    biasw = (float*)   (ws + 3 * sz_qkv);                               // 1.6MB
  char* p2 = ws + 3 * sz_qkv + (size_t)HEADS * NPAIR * sizeof(float);
  _Float16* aout  = (_Float16*)(p2);                                            // 52.4MB
  char* p3 = p2 + (size_t)NPAIR * INNER * sizeof(_Float16);
  _Float16* Wqkvt = (_Float16*)(p3);                                            // 192KB
  _Float16* Woutt = (_Float16*)(p3 + (size_t)QKVN * DIMD * sizeof(_Float16));   // 64KB

  // K0: weight transpose/convert (768*128 + 128*256 = 131072 elements)
  prep_weights<<<512, 256, 0, stream>>>(Wqkv, Wout, Wqkvt, Woutt);
  // K1: additive attention bias
  bias_kernel<<<NPAIR / 256, 256, 0, stream>>>(P, Wbias, biasw);
  // K2: QKV projection (M=102400 -> 1600 tiles of 64; N=768 -> 6 tiles of 128)
  qkv_kernel<<<dim3(NPAIR / 64, QKVN / 128), 256, 0, stream>>>(P, Wqkvt, qws, kws, vtws);
  // K3: attention: 320*4*20 = 25600 wave-tasks, 4 waves/block
  attn_kernel<<<25600 / 4, 128, 0, stream>>>(qws, kws, vtws, biasw, aout);
  // K4: output projection
  outproj_kernel<<<NPAIR / 64, 256, 0, stream>>>(aout, Woutt, bout, out);
}
